// MultiHeadAttention_70557722738877
// MI455X (gfx1250) — matrix operand order repaired and checked by reading, not compiled, not run on hardware
//
#include <hip/hip_runtime.h>

// MHA forward for MI455X (gfx1250): bf16 WMMA pipeline with fp32 accumulate,
// register-blocked GEMMs (16x64 per wave, loads hoisted for pipelining) and
// async-to-LDS K/V staging in the flash-attention kernel (double-buffered,
// ASYNCcnt-synchronized).
constexpr int Bc = 4, Sc = 2048, Dc = 1024, Hc = 16, DHc = 64;

typedef __bf16 bf16;
typedef __attribute__((ext_vector_type(16))) __bf16 v16bf;
typedef __attribute__((ext_vector_type(8)))  __bf16 v8bf;
typedef __attribute__((ext_vector_type(8)))  float  v8f;

__device__ __forceinline__ v16bf cat16(v8bf lo, v8bf hi) {
  v16bf r;
#pragma unroll
  for (int i = 0; i < 8; ++i) { r[i] = lo[i]; r[i + 8] = hi[i]; }
  return r;
}

// A-operand fragment (16x32, M=lane%16): 16 contiguous K-elements per lane
// (caller adds half*16).
__device__ __forceinline__ v16bf ldA(const bf16* p) {
  return cat16(*(const v8bf*)(p), *(const v8bf*)(p + 8));
}
// B-operand fragment (32x16, N=lane%16): 16 contiguous K-elements per lane
// (caller adds half*16).
__device__ __forceinline__ v16bf ldB(const bf16* p) {
  return cat16(*(const v8bf*)(p), *(const v8bf*)(p + 8));
}

__device__ __forceinline__ v8f wmma_bf16(v16bf a, v16bf b, v8f c) {
  return __builtin_amdgcn_wmma_f32_16x16x32_bf16(false, a, false, b, (short)0, c,
                                                 false, false);
}

// ---- CDNA5 async LDS fill: 16B per lane, tracked by ASYNCcnt -------------
__device__ __forceinline__ void async_g2l_b128(unsigned lds_off, uint64_t gaddr) {
  asm volatile("global_load_async_to_lds_b128 %0, %1, off"
               :: "v"(lds_off), "v"(gaddr) : "memory");
}
__device__ __forceinline__ void wait_async0() {
#if __has_builtin(__builtin_amdgcn_s_wait_asynccnt)
  __builtin_amdgcn_s_wait_asynccnt(0);
#else
  asm volatile("s_wait_asynccnt 0x0" ::: "memory");
#endif
}
__device__ __forceinline__ void wait_async2() {
#if __has_builtin(__builtin_amdgcn_s_wait_asynccnt)
  __builtin_amdgcn_s_wait_asynccnt(2);
#else
  asm volatile("s_wait_asynccnt 0x2" ::: "memory");
#endif
}

// ---------------- conversion / layout kernels ----------------

__global__ __launch_bounds__(256) void cvt_x(const float* __restrict__ in,
                                             bf16* __restrict__ out) {
  size_t i = (size_t)blockIdx.x * 256 + threadIdx.x;  // exact multiple
  out[i] = (bf16)in[i];
}

// Wq/Wk/Wv [H,D,DH] fp32 -> WT [3][H][DH][D] bf16 (K-contiguous B-operand).
__global__ __launch_bounds__(256) void cvt_wqkv(const float* __restrict__ Wq,
                                                const float* __restrict__ Wk,
                                                const float* __restrict__ Wv,
                                                bf16* __restrict__ WT) {
  size_t i = (size_t)blockIdx.x * 256 + threadIdx.x;  // 3*H*DH*D threads exact
  int which = (int)(i / ((size_t)Hc * DHc * Dc));
  size_t r = i % ((size_t)Hc * DHc * Dc);
  int h = (int)(r / (DHc * Dc));
  int r2 = (int)(r % (DHc * Dc));
  int e = r2 / Dc, d = r2 % Dc;
  const float* W = which == 0 ? Wq : (which == 1 ? Wk : Wv);
  WT[i] = (bf16)W[((size_t)h * Dc + d) * DHc + e];
}

// Wo [D,D] fp32 -> WoT [f][d] bf16.
__global__ __launch_bounds__(256) void cvt_wo(const float* __restrict__ Wo,
                                              bf16* __restrict__ WoT) {
  size_t i = (size_t)blockIdx.x * 256 + threadIdx.x;  // D*D threads exact
  int f = (int)(i / Dc), d = (int)(i % Dc);
  WoT[i] = (bf16)Wo[(size_t)d * Dc + f];
}

// ---------------- QKV projection ----------------
// One wave per 16-row strip x full DH(64) of one of {Q,K,V}: 4 accumulators,
// A-fragment reused 4x per k-step. All 5 fragment loads per k-step hoisted
// ahead of the 4 WMMAs so they clause together and waits retire partially.
// V stored transposed [B,H,DH,S].
__global__ __launch_bounds__(256) void qkv_gemm(const bf16* __restrict__ xb,
                                                const bf16* __restrict__ WT,
                                                bf16* __restrict__ Qb,
                                                bf16* __restrict__ Kb,
                                                bf16* __restrict__ Vt) {
  int w = blockIdx.x * 8 + (threadIdx.x >> 5);   // 24576 waves
  int l = threadIdx.x & 31, half = l >> 4, lm = l & 15;
  int which = w >> 13;          // 8192 = 2^13 strips per matrix
  int t = w & 8191;
  int s0 = (t & 127) * 16;
  int h  = (t >> 7) & 15;
  int b  = t >> 11;

  const bf16* arow  = xb + ((size_t)b * Sc + s0 + lm) * Dc;
  const bf16* wbase = WT + (size_t)which * Hc * DHc * Dc +
                      ((size_t)h * DHc + lm) * Dc;
  v8f c[4] = {};
#pragma unroll 2
  for (int d0 = 0; d0 < Dc; d0 += 32) {
    v16bf a  = ldA(arow + d0 + half * 16);
    v16bf b0 = ldB(wbase + (size_t)0 * 16 * Dc + d0 + half * 16);
    v16bf b1 = ldB(wbase + (size_t)1 * 16 * Dc + d0 + half * 16);
    v16bf b2 = ldB(wbase + (size_t)2 * 16 * Dc + d0 + half * 16);
    v16bf b3 = ldB(wbase + (size_t)3 * 16 * Dc + d0 + half * 16);
    c[0] = wmma_bf16(a, b0, c[0]);
    c[1] = wmma_bf16(a, b1, c[1]);
    c[2] = wmma_bf16(a, b2, c[2]);
    c[3] = wmma_bf16(a, b3, c[3]);
  }

  size_t bh = (size_t)b * Hc + h;
  if (which == 2) {
#pragma unroll
    for (int e = 0; e < 4; ++e)
#pragma unroll
      for (int i = 0; i < 8; ++i) {
        int s = s0 + i + half * 8;
        Vt[(bh * DHc + e * 16 + lm) * Sc + s] = (bf16)c[e][i];
      }
  } else {
    bf16* O = (which == 0) ? Qb : Kb;
#pragma unroll
    for (int e = 0; e < 4; ++e)
#pragma unroll
      for (int i = 0; i < 8; ++i) {
        int s = s0 + i + half * 8;
        O[(bh * Sc + s) * DHc + e * 16 + lm] = (bf16)c[e][i];
      }
  }
}

// ---------------- causal flash attention ----------------
// Block = 8 waves = 128 query rows of one (b,h). Per 32-key step, the block
// cooperatively async-stages the K tile (32x64) and V^T tile (64x32) into
// double-buffered LDS (one b128 async op per lane per tile), then each wave
// runs QK^T (4 WMMA) + online softmax + P*V (4 WMMA) from LDS.
__global__ __launch_bounds__(256) void attn_fwd(const bf16* __restrict__ Qb,
                                                const bf16* __restrict__ Kb,
                                                const bf16* __restrict__ Vt,
                                                bf16* __restrict__ concat) {
  __shared__ bf16 kbuf[2][32 * 64];   // 2 x 4KB
  __shared__ bf16 vbuf[2][64 * 32];   // 2 x 4KB
  __shared__ bf16 pbuf[8][16 * 32];   // 8KB per-wave P staging
  int tid = threadIdx.x;
  int wid = tid >> 5;
  int l = tid & 31, half = l >> 4, lm = l & 15;
  int cb = blockIdx.x & 15;           // S/128 = 16 chunks
  int h  = (blockIdx.x >> 4) & 15;
  int b  = blockIdx.x >> 8;
  int chunk0 = cb * 128;
  int q0 = chunk0 + wid * 16;
  size_t bh = (size_t)b * Hc + h;

  const bf16* Kbh = Kb + bh * Sc * DHc;
  const bf16* Vbh = Vt + bh * DHc * Sc;
  const bf16* qrow = Qb + (bh * Sc + q0 + lm) * DHc;
  v16bf qa0 = ldA(qrow + half * 16);        // DH 0..31
  v16bf qa1 = ldA(qrow + 32 + half * 16);   // DH 32..63
  bf16* pw = &pbuf[wid][0];

  // staging assignment: K tile 32 rows x 128B (8 segs); V tile 64 rows x 64B.
  int kr = tid >> 3, ks = tid & 7;
  int vr = tid >> 2, vs = tid & 3;

  v8f oc[4] = {};
  float m_run[8], l_run[8];
#pragma unroll
  for (int i = 0; i < 8; ++i) { m_run[i] = -1e30f; l_run[i] = 0.f; }

  const float scale = 0.125f;             // 1/sqrt(64)
  int nt = chunk0 / 32 + 4;               // key steps covering causal extent

  // prologue: stage step 0 into buffer 0
  {
    const bf16* kg = Kbh + (size_t)kr * DHc + ks * 8;
    async_g2l_b128((unsigned)(uintptr_t)&kbuf[0][kr * 64 + ks * 8],
                   (uint64_t)(uintptr_t)kg);
    const bf16* vg = Vbh + (size_t)vr * Sc + vs * 8;
    async_g2l_b128((unsigned)(uintptr_t)&vbuf[0][vr * 32 + vs * 8],
                   (uint64_t)(uintptr_t)vg);
  }

  for (int it = 0; it < nt; ++it) {
    int t0 = it * 32;
    if (it + 1 < nt) {                    // stage next step into other buffer
      int tn = t0 + 32, nb = (it + 1) & 1;
      const bf16* kg = Kbh + (size_t)(tn + kr) * DHc + ks * 8;
      async_g2l_b128((unsigned)(uintptr_t)&kbuf[nb][kr * 64 + ks * 8],
                     (uint64_t)(uintptr_t)kg);
      const bf16* vg = Vbh + (size_t)vr * Sc + tn + vs * 8;
      async_g2l_b128((unsigned)(uintptr_t)&vbuf[nb][vr * 32 + vs * 8],
                     (uint64_t)(uintptr_t)vg);
      wait_async2();                      // this step's 2 ops done, next 2 fly
    } else {
      wait_async0();
    }
    __syncthreads();                      // all waves' staging visible

    if (t0 < q0 + 16) {                   // causally relevant for this wave
      const bf16* kb_ = &kbuf[it & 1][0];
      const bf16* vb_ = &vbuf[it & 1][0];
      const bf16* k0p = kb_ + (size_t)lm * 64 + half * 16;
      const bf16* k1p = kb_ + (size_t)(16 + lm) * 64 + half * 16;

      v8f c0 = {}, c1 = {};
      c0 = wmma_bf16(qa0, ldB(k0p), c0);
      c0 = wmma_bf16(qa1, ldB(k0p + 32), c0);
      c1 = wmma_bf16(qa0, ldB(k1p), c1);
      c1 = wmma_bf16(qa1, ldB(k1p + 32), c1);

      float corr[8];
#pragma unroll
      for (int i = 0; i < 8; ++i) {
        int q = q0 + i + half * 8;
        float s0v = c0[i] * scale; if (t0 + lm > q)      s0v = -1e30f;
        float s1v = c1[i] * scale; if (t0 + 16 + lm > q) s1v = -1e30f;
        float rm = fmaxf(s0v, s1v);
        rm = fmaxf(rm, __shfl_xor(rm, 1, 32));
        rm = fmaxf(rm, __shfl_xor(rm, 2, 32));
        rm = fmaxf(rm, __shfl_xor(rm, 4, 32));
        rm = fmaxf(rm, __shfl_xor(rm, 8, 32));
        float newm = fmaxf(m_run[i], rm);
        float p0 = __expf(s0v - newm);
        float p1 = __expf(s1v - newm);
        float rs = p0 + p1;
        rs += __shfl_xor(rs, 1, 32);
        rs += __shfl_xor(rs, 2, 32);
        rs += __shfl_xor(rs, 4, 32);
        rs += __shfl_xor(rs, 8, 32);
        float cr = __expf(m_run[i] - newm);
        l_run[i] = l_run[i] * cr + rs;
        m_run[i] = newm;
        corr[i] = cr;
        int row = i + half * 8;           // C-layout -> row-major P in LDS
        pw[row * 32 + lm]      = (bf16)p0;
        pw[row * 32 + 16 + lm] = (bf16)p1;
      }
#pragma unroll
      for (int e = 0; e < 4; ++e)
#pragma unroll
        for (int i = 0; i < 8; ++i) oc[e][i] *= corr[i];

      v16bf pa = ldA(pw + lm * 32 + half * 16);   // P as A-operand (16x32)
#pragma unroll
      for (int e = 0; e < 4; ++e) {
        const bf16* vp = vb_ + (size_t)(e * 16 + lm) * 32 + half * 16;
        oc[e] = wmma_bf16(pa, ldB(vp), oc[e]);
      }
    }
    __syncthreads();                      // reads done before buffer reuse
  }

  float inv[8];
#pragma unroll
  for (int i = 0; i < 8; ++i) inv[i] = (l_run[i] > 0.f) ? 1.f / l_run[i] : 0.f;
#pragma unroll
  for (int e = 0; e < 4; ++e)
#pragma unroll
    for (int i = 0; i < 8; ++i) {
      int s = q0 + i + half * 8;
      concat[((size_t)b * Sc + s) * Dc + h * 64 + e * 16 + lm] =
          (bf16)(oc[e][i] * inv[i]);
    }
}

// ---------------- output projection + bias ----------------
// One wave per 16-row x 64-col strip; loads hoisted as in qkv_gemm.
__global__ __launch_bounds__(256) void out_proj(const bf16* __restrict__ cc,
                                                const bf16* __restrict__ WoT,
                                                const float* __restrict__ bo,
                                                float* __restrict__ out) {
  int w = blockIdx.x * 8 + (threadIdx.x >> 5);   // 8192 waves
  int l = threadIdx.x & 31, half = l >> 4, lm = l & 15;
  int f0 = (w & 15) * 64;
  int r0 = (w >> 4) * 16;     // row in flattened [B*S]

  const bf16* arow = cc + (size_t)(r0 + lm) * Dc;
  const bf16* brow = WoT + (size_t)(f0 + lm) * Dc;
  v8f c[4] = {};
#pragma unroll 2
  for (int d0 = 0; d0 < Dc; d0 += 32) {
    v16bf a  = ldA(arow + d0 + half * 16);
    v16bf b0 = ldB(brow + (size_t)0 * 16 * Dc + d0 + half * 16);
    v16bf b1 = ldB(brow + (size_t)1 * 16 * Dc + d0 + half * 16);
    v16bf b2 = ldB(brow + (size_t)2 * 16 * Dc + d0 + half * 16);
    v16bf b3 = ldB(brow + (size_t)3 * 16 * Dc + d0 + half * 16);
    c[0] = wmma_bf16(a, b0, c[0]);
    c[1] = wmma_bf16(a, b1, c[1]);
    c[2] = wmma_bf16(a, b2, c[2]);
    c[3] = wmma_bf16(a, b3, c[3]);
  }
#pragma unroll
  for (int e = 0; e < 4; ++e) {
    float bias = bo[f0 + e * 16 + lm];
#pragma unroll
    for (int i = 0; i < 8; ++i)
      out[(size_t)(r0 + i + half * 8) * Dc + f0 + e * 16 + lm] = c[e][i] + bias;
  }
}

extern "C" void kernel_launch(void* const* d_in, const int* in_sizes, int n_in,
                              void* d_out, int out_size, void* d_ws, size_t ws_size,
                              hipStream_t stream) {
  const float* x  = (const float*)d_in[0];
  const float* Wq = (const float*)d_in[1];
  const float* Wk = (const float*)d_in[2];
  const float* Wv = (const float*)d_in[3];
  const float* Wo = (const float*)d_in[4];
  const float* bo = (const float*)d_in[5];
  float* out = (float*)d_out;

  bf16* p = (bf16*)d_ws;
  bf16* xb  = p; p += (size_t)Bc * Sc * Dc;            // 8 Mi elem
  bf16* WT  = p; p += (size_t)3 * Hc * DHc * Dc;       // 3 Mi
  bf16* WoT = p; p += (size_t)Dc * Dc;                 // 1 Mi
  bf16* Qb  = p; p += (size_t)Bc * Hc * Sc * DHc;      // 8 Mi
  bf16* Kb  = p; p += (size_t)Bc * Hc * Sc * DHc;      // 8 Mi
  bf16* Vt  = p; p += (size_t)Bc * Hc * DHc * Sc;      // 8 Mi
  bf16* cc  = p;                                       // 8 Mi -> ~88 MB total

  cvt_x<<<(Bc * Sc * Dc) / 256, 256, 0, stream>>>(x, xb);
  cvt_wqkv<<<(3 * Hc * DHc * Dc) / 256, 256, 0, stream>>>(Wq, Wk, Wv, WT);
  cvt_wo<<<(Dc * Dc) / 256, 256, 0, stream>>>(Wo, WoT);

  // 3 * B*H*(S/16) = 24576 waves, 8 per block.
  qkv_gemm<<<24576 / 8, 256, 0, stream>>>(xb, WT, Qb, Kb, Vt);
  // B*H*(S/128) = 1024 blocks of 8 waves (128 query rows each).
  attn_fwd<<<1024, 256, 0, stream>>>(Qb, Kb, Vt, cc);
  // (B*S/16)*(D/64) = 8192 waves.
  out_proj<<<8192 / 8, 256, 0, stream>>>(cc, WoT, bo, out);
}
